// HAN_26740466385345
// MI455X (gfx1250) — compile-verified, matrix operand order checked by reading, not run on hardware
//
#include <hip/hip_runtime.h>
#include <hip/hip_bf16.h>

typedef __attribute__((ext_vector_type(16))) _Float16 v16h;
typedef __attribute__((ext_vector_type(8)))  _Float16 v8h;
typedef __attribute__((ext_vector_type(8)))  float    v8f;
typedef _Float16 f16;

// ---------------------------------------------------------------------------
// WMMA GEMM:  C[M,N] = act( A[M,K](f16) * B(f16) + bias[N] ),  C is f32.
// bTrans==0: B is [K,N] row-major (ldb = row stride).
// bTrans==1: B is [N,K] row-major (computes A @ B^T).
// act: 0=none, 1=elu, 2=relu (== relu(elu(x)))
// Block: 128 threads = 4 waves; 64x64 output tile; each wave does 2x2 WMMA.
// LDS: A tile [BM][BK], B tile stored TRANSPOSED [BN][BK] so both fragments
// are two contiguous 16B chunks per lane -> ds_load_b128.
// ---------------------------------------------------------------------------
#define BM 64
#define BN 64
#define BK 32
#define LDS_T 40   // BK + 8 halves pad (80B row stride, 16B-aligned)

__global__ __launch_bounds__(128)
void gemm_wmma(const f16* __restrict__ A, int lda,
               const f16* __restrict__ B, int ldb, int bTrans,
               float* __restrict__ C, int ldc,
               int M, int N, int K,
               const float* __restrict__ bias, int act)
{
    __shared__ __align__(16) f16 As[BM * LDS_T];
    __shared__ __align__(16) f16 Bs[BN * LDS_T];   // transposed: [n][k]

    const int tid  = threadIdx.x;
    const int lane = tid & 31;
    const int wave = tid >> 5;
    const int g    = lane >> 4;     // lane half-group (0/1)
    const int lm   = lane & 15;

    const int blockRow = blockIdx.y * BM;
    const int blockCol = blockIdx.x * BN;
    const int waveRow  = (wave >> 1) * 32;
    const int waveCol  = (wave & 1) * 32;

    v8f acc[2][2];
#pragma unroll
    for (int i = 0; i < 2; ++i)
#pragma unroll
        for (int j = 0; j < 2; ++j)
#pragma unroll
            for (int e = 0; e < 8; ++e) acc[i][j][e] = 0.0f;

    const bool rowsFull = (blockRow + BM <= M);
    const bool colsFull = (blockCol + BN <= N);
    const bool aAligned = ((lda & 7) == 0);
    const bool bAligned = ((ldb & 7) == 0);

    // cooperative loader indices
    const int aRow = tid >> 1;           // 0..63
    const int aCol = (tid & 1) * 16;     // 0 / 16
    const int bK   = tid >> 2;           // 0..31  (bTrans==0 path)
    const int bNc  = (tid & 3) * 16;     // 0/16/32/48

    for (int k0 = 0; k0 < K; k0 += BK) {
        const bool kFull = (k0 + BK <= K);

        // ---- stage A tile [BM][BK] ----
        if (rowsFull && kFull && aAligned) {
            const uint4* src = (const uint4*)(A + (size_t)(blockRow + aRow) * lda + k0 + aCol);
            uint4 q0 = src[0], q1 = src[1];
            *(uint4*)&As[aRow * LDS_T + aCol]     = q0;
            *(uint4*)&As[aRow * LDS_T + aCol + 8] = q1;
        } else {
            int gRow = blockRow + aRow;
#pragma unroll 1
            for (int i = 0; i < 16; ++i) {
                int gk = k0 + aCol + i;
                f16 v = (f16)0.0f;
                if (gRow < M && gk < K) v = A[(size_t)gRow * lda + gk];
                As[aRow * LDS_T + aCol + i] = v;
            }
        }

        // ---- stage B tile -> Bs[n][k] (transposed) ----
        if (bTrans) {
            // B is [N,K]: rows are K-contiguous, natural for transposed tile
            if (colsFull && kFull && bAligned) {
                const uint4* src = (const uint4*)(B + (size_t)(blockCol + aRow) * ldb + k0 + aCol);
                uint4 q0 = src[0], q1 = src[1];
                *(uint4*)&Bs[aRow * LDS_T + aCol]     = q0;
                *(uint4*)&Bs[aRow * LDS_T + aCol + 8] = q1;
            } else {
                int gn = blockCol + aRow;
#pragma unroll 1
                for (int i = 0; i < 16; ++i) {
                    int gk = k0 + aCol + i;
                    f16 v = (f16)0.0f;
                    if (gn < N && gk < K) v = B[(size_t)gn * ldb + gk];
                    Bs[aRow * LDS_T + aCol + i] = v;
                }
            }
        } else {
            // B is [K,N]: read rows vectorized, scatter-transpose into LDS
            if (colsFull && kFull && bAligned) {
                union { uint4 q[2]; f16 h[16]; } u;
                const uint4* src = (const uint4*)(B + (size_t)(k0 + bK) * ldb + blockCol + bNc);
                u.q[0] = src[0]; u.q[1] = src[1];
#pragma unroll
                for (int i = 0; i < 16; ++i)
                    Bs[(bNc + i) * LDS_T + bK] = u.h[i];
            } else {
                int gk = k0 + bK;
#pragma unroll 1
                for (int i = 0; i < 16; ++i) {
                    int gn = blockCol + bNc + i;
                    f16 v = (f16)0.0f;
                    if (gk < K && gn < N) v = B[(size_t)gk * ldb + gn];
                    Bs[(bNc + i) * LDS_T + bK] = v;
                }
            }
        }
        __syncthreads();

        // prefetch next A tile (global_prefetch_b8)
        if (k0 + BK < K && rowsFull && aAligned)
            __builtin_prefetch(A + (size_t)(blockRow + aRow) * lda + (k0 + BK) + aCol, 0, 1);

        // ---- fragments: two 16B LDS chunks each (halves 8g..+7, 16+8g..+7) ----
        v16h afrag[2], bfrag[2];
#pragma unroll
        for (int t = 0; t < 2; ++t) {
            int m = waveRow + t * 16 + lm;
            const v8h a0 = *(const v8h*)&As[m * LDS_T + 8 * g];
            const v8h a1 = *(const v8h*)&As[m * LDS_T + 16 + 8 * g];
            afrag[t] = __builtin_shufflevector(a0, a1,
                0,1,2,3,4,5,6,7,8,9,10,11,12,13,14,15);
        }
#pragma unroll
        for (int t = 0; t < 2; ++t) {
            int n = waveCol + t * 16 + lm;
            const v8h b0 = *(const v8h*)&Bs[n * LDS_T + 8 * g];
            const v8h b1 = *(const v8h*)&Bs[n * LDS_T + 16 + 8 * g];
            bfrag[t] = __builtin_shufflevector(b0, b1,
                0,1,2,3,4,5,6,7,8,9,10,11,12,13,14,15);
        }

#pragma unroll
        for (int i = 0; i < 2; ++i)
#pragma unroll
            for (int j = 0; j < 2; ++j)
                acc[i][j] = __builtin_amdgcn_wmma_f32_16x16x32_f16(
                    false, afrag[i], false, bfrag[j],
                    (short)0, acc[i][j], false, false);

        __syncthreads();
    }

    // ---- epilogue: C layout M = r + 8*g, N = lane%16 ----
#pragma unroll
    for (int i = 0; i < 2; ++i) {
#pragma unroll
        for (int j = 0; j < 2; ++j) {
            int gn = blockCol + waveCol + j * 16 + lm;
            float bv = (bias != nullptr && gn < N) ? bias[gn] : 0.0f;
#pragma unroll
            for (int r = 0; r < 8; ++r) {
                int gm = blockRow + waveRow + i * 16 + r + 8 * g;
                if (gm < M && gn < N) {
                    float v = acc[i][j][r] + bv;
                    if (act == 1)      v = v > 0.0f ? v : (__expf(v) - 1.0f);
                    else if (act == 2) v = v > 0.0f ? v : 0.0f;
                    C[(size_t)gm * ldc + gn] = v;
                }
            }
        }
    }
}

// ---------------------------------------------------------------------------
// Elementwise / reduction helpers
// ---------------------------------------------------------------------------
__global__ void cvt_f16_kernel(const float* __restrict__ in, f16* __restrict__ out, size_t cnt) {
    size_t i  = ((size_t)blockIdx.x * blockDim.x + threadIdx.x) * 4;
    size_t st = (size_t)gridDim.x * blockDim.x * 4;
    for (; i + 3 < cnt; i += st) {
        float4 v = *(const float4*)(in + i);
        union { f16 h[4]; uint2 u; } o;
        o.h[0] = (f16)v.x; o.h[1] = (f16)v.y; o.h[2] = (f16)v.z; o.h[3] = (f16)v.w;
        *(uint2*)(out + i) = o.u;
    }
    if (blockIdx.x == 0 && threadIdx.x == 0)
        for (size_t j = cnt & ~(size_t)3; j < cnt; ++j) out[j] = (f16)in[j];
}

__global__ void zero_f16_kernel(f16* __restrict__ out, size_t cnt) {
    size_t i  = ((size_t)blockIdx.x * blockDim.x + threadIdx.x) * 4;
    size_t st = (size_t)gridDim.x * blockDim.x * 4;
    uint2 z; z.x = 0u; z.y = 0u;
    for (; i + 3 < cnt; i += st) *(uint2*)(out + i) = z;
    if (blockIdx.x == 0 && threadIdx.x == 0)
        for (size_t j = cnt & ~(size_t)3; j < cnt; ++j) out[j] = (f16)0.0f;
}

__global__ __launch_bounds__(256)
void l2norm_rows_f16(const float* __restrict__ in, f16* __restrict__ out, int d) {
    __shared__ float red[256];
    int row = blockIdx.x;
    float s = 0.0f;
    for (int j = threadIdx.x; j < d; j += 256) {
        float v = in[(size_t)row * d + j];
        s += v * v;
    }
    red[threadIdx.x] = s; __syncthreads();
    for (int st = 128; st > 0; st >>= 1) {
        if (threadIdx.x < st) red[threadIdx.x] += red[threadIdx.x + st];
        __syncthreads();
    }
    float inv = 1.0f / fmaxf(sqrtf(red[0]), 1e-12f);
    for (int j = threadIdx.x; j < d; j += 256)
        out[(size_t)row * d + j] = (f16)(in[(size_t)row * d + j] * inv);
}

// per-row: proj[row] = sum_h tanh(t[row][h]) * wb[h]   (bias already fused)
__global__ __launch_bounds__(128)
void rowdot_tanh(const float* __restrict__ t, const float* __restrict__ wb,
                 float* __restrict__ proj, int d) {
    __shared__ float red[128];
    int row = blockIdx.x;
    float s = 0.0f;
    for (int j = threadIdx.x; j < d; j += 128)
        s += tanhf(t[(size_t)row * d + j]) * wb[j];
    red[threadIdx.x] = s; __syncthreads();
    for (int st = 64; st > 0; st >>= 1) {
        if (threadIdx.x < st) red[threadIdx.x] += red[threadIdx.x + st];
        __syncthreads();
    }
    if (threadIdx.x == 0) proj[row] = red[0];
}

// beta = softmax(mean over n of proj[m][:]) for m=0..2
__global__ __launch_bounds__(256)
void mean_softmax3(const float* __restrict__ proj, int n, float* __restrict__ beta) {
    __shared__ float red[256];
    float means[3];
    for (int m = 0; m < 3; ++m) {
        float s = 0.0f;
        for (int j = threadIdx.x; j < n; j += 256) s += proj[(size_t)m * n + j];
        red[threadIdx.x] = s; __syncthreads();
        for (int st = 128; st > 0; st >>= 1) {
            if (threadIdx.x < st) red[threadIdx.x] += red[threadIdx.x + st];
            __syncthreads();
        }
        means[m] = red[0] / (float)n;
        __syncthreads();
    }
    if (threadIdx.x == 0) {
        float mx = fmaxf(means[0], fmaxf(means[1], means[2]));
        float e0 = expf(means[0] - mx), e1 = expf(means[1] - mx), e2 = expf(means[2] - mx);
        float s = e0 + e1 + e2;
        beta[0] = e0 / s; beta[1] = e1 / s; beta[2] = e2 / s;
    }
}

// x = sum_m beta[m] * z[m];  also write f16 copy
__global__ void weighted_sum3(const float* __restrict__ z, const float* __restrict__ beta,
                              float* __restrict__ x, f16* __restrict__ xh,
                              size_t cnt, size_t stride) {
    size_t i = (size_t)blockIdx.x * blockDim.x + threadIdx.x;
    if (i < cnt) {
        float v = beta[0] * z[i] + beta[1] * z[i + stride] + beta[2] * z[i + 2 * stride];
        x[i] = v;
        xh[i] = (f16)v;
    }
}

// feature[i] = concat(x1[data[i,0]], x2[data[i,1]])
__global__ __launch_bounds__(128)
void build_feature(const float* __restrict__ x1, const float* __restrict__ x2,
                   const int* __restrict__ data, float* __restrict__ feat) {
    int i = blockIdx.x;
    int r1 = data[2 * i], r2 = data[2 * i + 1];
    int j = threadIdx.x;   // blockDim == 128
    feat[(size_t)i * 256 + j]       = x1[(size_t)r1 * 128 + j];
    feat[(size_t)i * 256 + 128 + j] = x2[(size_t)r2 * 128 + j];
}

// top-k selection per row: set fedge=1, sedge=edge at the k largest sim entries
__global__ __launch_bounds__(256)
void topk_kernel(float* __restrict__ sim, const float* __restrict__ edge,
                 f16* __restrict__ fedge, f16* __restrict__ sedge, int n, int k) {
    __shared__ float sv[256];
    __shared__ int   si[256];
    int row = blockIdx.x;
    float* srow = sim + (size_t)row * n;
    for (int it = 0; it < k; ++it) {
        float best = -3.4e38f; int bidx = 0;
        for (int j = threadIdx.x; j < n; j += 256) {
            float v = srow[j];
            if (v > best) { best = v; bidx = j; }
        }
        sv[threadIdx.x] = best; si[threadIdx.x] = bidx;
        __syncthreads();
        for (int s = 128; s > 0; s >>= 1) {
            if (threadIdx.x < s && sv[threadIdx.x + s] > sv[threadIdx.x]) {
                sv[threadIdx.x] = sv[threadIdx.x + s];
                si[threadIdx.x] = si[threadIdx.x + s];
            }
            __syncthreads();
        }
        if (threadIdx.x == 0) {
            int idx = si[0];
            srow[idx] = -3.4e38f;
            fedge[(size_t)row * n + idx] = (f16)1.0f;
            sedge[(size_t)row * n + idx] = (f16)edge[(size_t)row * n + idx];
        }
        __threadfence_block();
        __syncthreads();
    }
}

__global__ __launch_bounds__(256)
void gather_rows_f16(const float* __restrict__ emb, const int* __restrict__ idx,
                     f16* __restrict__ out, int d) {
    int i = blockIdx.x;
    int r = idx[i];
    for (int j = threadIdx.x; j < d; j += 256)
        out[(size_t)i * d + j] = (f16)emb[(size_t)r * d + j];
}

// logits = hid @ W2^T (2x128), then row log_softmax -> out
__global__ void mlp2_logsoftmax(const float* __restrict__ hid, const float* __restrict__ W2,
                                float* __restrict__ out, int rows) {
    int i = blockIdx.x * blockDim.x + threadIdx.x;
    if (i >= rows) return;
    float a = 0.0f, b = 0.0f;
    for (int kk = 0; kk < 128; ++kk) {
        float h = hid[(size_t)i * 128 + kk];
        a += h * W2[kk];
        b += h * W2[128 + kk];
    }
    float mx  = fmaxf(a, b);
    float lse = mx + logf(expf(a - mx) + expf(b - mx));
    out[2 * i]     = a - lse;
    out[2 * i + 1] = b - lse;
}

// ---------------------------------------------------------------------------
// Host-side orchestration
// ---------------------------------------------------------------------------
static inline char* bump(char*& p, size_t bytes) {
    char* r = p;
    p += (bytes + 255) & ~(size_t)255;
    return r;
}

static void gemm(hipStream_t s, const f16* A, int lda, const f16* B, int ldb, int bT,
                 float* C, int ldc, int M, int N, int K, const float* bias, int act) {
    dim3 grid((N + BN - 1) / BN, (M + BM - 1) / BM);
    gemm_wmma<<<grid, 128, 0, s>>>(A, lda, B, ldb, bT, C, ldc, M, N, K, bias, act);
}

static void cvt(hipStream_t s, const float* in, f16* out, size_t cnt) {
    size_t b = (cnt / 4 + 255) / 256; if (b > 4096) b = 4096; if (b == 0) b = 1;
    cvt_f16_kernel<<<(unsigned)b, 256, 0, s>>>(in, out, cnt);
}

static void zero16(hipStream_t s, f16* out, size_t cnt) {
    size_t b = (cnt / 4 + 255) / 256; if (b > 4096) b = 4096; if (b == 0) b = 1;
    zero_f16_kernel<<<(unsigned)b, 256, 0, s>>>(out, cnt);
}

struct HanBufs {
    float *x, *sup, *z, *tmp, *proj, *beta;
    f16 *xh, *suph, *zh;
};

static void run_han(hipStream_t stream, int n,
                    const f16* adjh,      // [3,n,n]
                    const f16* hinh,      // [n,512]
                    const f16* wtransh,   // [128,512]
                    const f16* gcWh,      // [2,3,128,128]
                    const float* gcb,     // [2,3,128]
                    const f16* saWah,     // [2,128,128]
                    const float* saba,    // [2,128]
                    const float* sawb,    // [2,128]
                    const HanBufs& b) {
    // x = h_in @ w_trans^T
    gemm(stream, hinh, 512, wtransh, 512, 1, b.x, 128, n, 128, 512, nullptr, 0);
    cvt(stream, b.x, b.xh, (size_t)n * 128);

    for (int hd = 0; hd < 2; ++hd) {
        for (int m = 0; m < 3; ++m)
            gemm(stream, b.xh, 128, gcWh + (size_t)(hd * 3 + m) * 128 * 128, 128, 0,
                 b.sup + (size_t)m * n * 128, 128, n, 128, 128, nullptr, 0);
        cvt(stream, b.sup, b.suph, (size_t)3 * n * 128);
        for (int m = 0; m < 3; ++m)
            gemm(stream, adjh + (size_t)m * n * n, n,
                 b.suph + (size_t)m * n * 128, 128, 0,
                 b.z + (size_t)m * n * 128, 128, n, 128, n,
                 gcb + (size_t)(hd * 3 + m) * 128, 1 /*elu*/);
        cvt(stream, b.z, b.zh, (size_t)3 * n * 128);
        for (int m = 0; m < 3; ++m) {
            gemm(stream, b.zh + (size_t)m * n * 128, 128,
                 saWah + (size_t)hd * 128 * 128, 128, 1 /*B^T*/,
                 b.tmp, 128, n, 128, 128, saba + (size_t)hd * 128, 0);
            rowdot_tanh<<<n, 128, 0, stream>>>(b.tmp, sawb + (size_t)hd * 128,
                                               b.proj + (size_t)m * n, 128);
        }
        mean_softmax3<<<1, 256, 0, stream>>>(b.proj, n, b.beta);
        size_t cnt = (size_t)n * 128;
        weighted_sum3<<<(unsigned)((cnt + 255) / 256), 256, 0, stream>>>(
            b.z, b.beta, b.x, b.xh, cnt, cnt);
    }
}

extern "C" void kernel_launch(void* const* d_in, const int* in_sizes, int n_in,
                              void* d_out, int out_size, void* d_ws, size_t ws_size,
                              hipStream_t stream) {
    (void)in_sizes; (void)n_in; (void)out_size; (void)ws_size;
    const int N1 = 3000, N2 = 4000, NP = 6000, NSEL = 2048, K = 15;

    const float* g1      = (const float*)d_in[0];
    const float* g2      = (const float*)d_in[1];
    const float* h1      = (const float*)d_in[2];
    const float* h2      = (const float*)d_in[3];
    const float* edge    = (const float*)d_in[4];
    const int*   data    = (const int*)d_in[5];
    const int*   dsel    = (const int*)d_in[6];
    const float* wt1     = (const float*)d_in[7];
    const float* wt2     = (const float*)d_in[8];
    const float* gcW1    = (const float*)d_in[9];
    const float* gcb1    = (const float*)d_in[10];
    const float* saWa1   = (const float*)d_in[11];
    const float* saba1   = (const float*)d_in[12];
    const float* sawb1   = (const float*)d_in[13];
    const float* gcW2    = (const float*)d_in[14];
    const float* gcb2    = (const float*)d_in[15];
    const float* saWa2   = (const float*)d_in[16];
    const float* saba2   = (const float*)d_in[17];
    const float* sawb2   = (const float*)d_in[18];
    const float* clW1    = (const float*)d_in[19];
    const float* clb1    = (const float*)d_in[20];
    const float* clW2    = (const float*)d_in[21];
    const float* clb2    = (const float*)d_in[22];
    const float* mlpW1   = (const float*)d_in[23];
    const float* mlpW2   = (const float*)d_in[24];
    float* out = (float*)d_out;

    // ----- workspace layout -----
    char* p = (char*)d_ws;
    f16* g1h    = (f16*)bump(p, (size_t)3 * N1 * N1 * 2);
    f16* g2h    = (f16*)bump(p, (size_t)3 * N2 * N2 * 2);
    f16* edgeh  = (f16*)bump(p, (size_t)NP * NP * 2);
    f16* fedgeh = (f16*)bump(p, (size_t)NP * NP * 2);
    f16* sedgeh = (f16*)bump(p, (size_t)NP * NP * 2);
    float* sim  = (float*)bump(p, (size_t)NP * NP * 4);

    f16* h1h   = (f16*)bump(p, (size_t)N1 * 512 * 2);
    f16* h2h   = (f16*)bump(p, (size_t)N2 * 512 * 2);
    f16* wt1h  = (f16*)bump(p, (size_t)128 * 512 * 2);
    f16* wt2h  = (f16*)bump(p, (size_t)128 * 512 * 2);
    f16* gcW1h = (f16*)bump(p, (size_t)2 * 3 * 128 * 128 * 2);
    f16* gcW2h = (f16*)bump(p, (size_t)2 * 3 * 128 * 128 * 2);
    f16* saWa1h= (f16*)bump(p, (size_t)2 * 128 * 128 * 2);
    f16* saWa2h= (f16*)bump(p, (size_t)2 * 128 * 128 * 2);
    f16* clW1h = (f16*)bump(p, (size_t)3 * 256 * 256 * 2);
    f16* clW2h = (f16*)bump(p, (size_t)3 * 256 * 256 * 2);
    f16* mlpW1h= (f16*)bump(p, (size_t)128 * 768 * 2);

    float* x1  = (float*)bump(p, (size_t)N1 * 128 * 4);
    float* x2  = (float*)bump(p, (size_t)N2 * 128 * 4);
    f16* x1h   = (f16*)bump(p, (size_t)N1 * 128 * 2);
    f16* x2h   = (f16*)bump(p, (size_t)N2 * 128 * 2);
    float* sup = (float*)bump(p, (size_t)3 * N2 * 128 * 4);
    f16* suph  = (f16*)bump(p, (size_t)3 * N2 * 128 * 2);
    float* zb  = (float*)bump(p, (size_t)3 * N2 * 128 * 4);
    f16* zh    = (f16*)bump(p, (size_t)3 * N2 * 128 * 2);
    float* tmp = (float*)bump(p, (size_t)N2 * 128 * 4);
    float* proj= (float*)bump(p, (size_t)3 * N2 * 4);
    float* beta= (float*)bump(p, 16 * 4);

    float* feat = (float*)bump(p, (size_t)NP * 256 * 4);
    f16* fnh    = (f16*)bump(p, (size_t)NP * 256 * 2);
    float* bufA = (float*)bump(p, (size_t)NP * 256 * 4);
    f16* bufAh  = (f16*)bump(p, (size_t)NP * 256 * 2);
    float* bufB = (float*)bump(p, (size_t)NP * 256 * 4);
    f16* bufBh  = (f16*)bump(p, (size_t)NP * 256 * 2);
    float* emb  = (float*)bump(p, (size_t)NP * 768 * 4);
    f16* selh   = (f16*)bump(p, (size_t)NSEL * 768 * 2);
    float* hid  = (float*)bump(p, (size_t)NSEL * 128 * 4);

    // ----- convert inputs/weights to f16 -----
    cvt(stream, g1, g1h, (size_t)3 * N1 * N1);
    cvt(stream, g2, g2h, (size_t)3 * N2 * N2);
    cvt(stream, edge, edgeh, (size_t)NP * NP);
    cvt(stream, h1, h1h, (size_t)N1 * 512);
    cvt(stream, h2, h2h, (size_t)N2 * 512);
    cvt(stream, wt1, wt1h, (size_t)128 * 512);
    cvt(stream, wt2, wt2h, (size_t)128 * 512);
    cvt(stream, gcW1, gcW1h, (size_t)2 * 3 * 128 * 128);
    cvt(stream, gcW2, gcW2h, (size_t)2 * 3 * 128 * 128);
    cvt(stream, saWa1, saWa1h, (size_t)2 * 128 * 128);
    cvt(stream, saWa2, saWa2h, (size_t)2 * 128 * 128);
    cvt(stream, clW1, clW1h, (size_t)3 * 256 * 256);
    cvt(stream, clW2, clW2h, (size_t)3 * 256 * 256);
    cvt(stream, mlpW1, mlpW1h, (size_t)128 * 768);

    // ----- HAN on both graphs -----
    HanBufs hb1 = { x1, sup, zb, tmp, proj, beta, x1h, suph, zh };
    run_han(stream, N1, g1h, h1h, wt1h, gcW1h, gcb1, saWa1h, saba1, sawb1, hb1);
    HanBufs hb2 = { x2, sup, zb, tmp, proj, beta, x2h, suph, zh };
    run_han(stream, N2, g2h, h2h, wt2h, gcW2h, gcb2, saWa2h, saba2, sawb2, hb2);

    // ----- pair features + kNN graph -----
    build_feature<<<NP, 128, 0, stream>>>(x1, x2, data, feat);
    l2norm_rows_f16<<<NP, 256, 0, stream>>>(feat, fnh, 256);
    gemm(stream, fnh, 256, fnh, 256, 1, sim, NP, NP, NP, 256, nullptr, 0);
    zero16(stream, fedgeh, (size_t)NP * NP);
    zero16(stream, sedgeh, (size_t)NP * NP);
    topk_kernel<<<NP, 256, 0, stream>>>(sim, edge, fedgeh, sedgeh, NP, K);

    // ----- 3 GCNs (shared normalized input fnh) -----
    for (int gI = 0; gI < 3; ++gI) {
        const f16* adjg = (gI == 0) ? edgeh : (gI == 1) ? fedgeh : sedgeh;
        gemm(stream, fnh, 256, clW1h + (size_t)gI * 256 * 256, 256, 0,
             bufA, 256, NP, 256, 256, nullptr, 0);
        cvt(stream, bufA, bufAh, (size_t)NP * 256);
        gemm(stream, adjg, NP, bufAh, 256, 0, bufB, 256, NP, 256, NP,
             clb1 + (size_t)gI * 256, 2 /*relu(elu)==relu*/);
        cvt(stream, bufB, bufBh, (size_t)NP * 256);
        gemm(stream, bufBh, 256, clW2h + (size_t)gI * 256 * 256, 256, 0,
             bufA, 256, NP, 256, 256, nullptr, 0);
        cvt(stream, bufA, bufAh, (size_t)NP * 256);
        gemm(stream, adjg, NP, bufAh, 256, 0, emb + (size_t)gI * 256, 768,
             NP, 256, NP, clb2 + (size_t)gI * 256, 1 /*elu*/);
    }

    // ----- MLP head -----
    gather_rows_f16<<<NSEL, 256, 0, stream>>>(emb, dsel, selh, 768);
    gemm(stream, selh, 768, mlpW1h, 768, 1, hid, 128, NSEL, 128, 768, nullptr, 1 /*elu*/);
    mlp2_logsoftmax<<<(NSEL + 127) / 128, 128, 0, stream>>>(hid, mlpW2, out, NSEL);

    // ----- outputs: pred, x1, x2 (flat concat) -----
    hipMemcpyAsync(out + (size_t)NSEL * 2, x1, (size_t)N1 * 128 * sizeof(float),
                   hipMemcpyDeviceToDevice, stream);
    hipMemcpyAsync(out + (size_t)NSEL * 2 + (size_t)N1 * 128, x2,
                   (size_t)N2 * 128 * sizeof(float), hipMemcpyDeviceToDevice, stream);
}
